// ECLGCNN_77902116815493
// MI455X (gfx1250) — matrix operand order repaired and checked by reading, not compiled, not run on hardware
//
#include <hip/hip_runtime.h>
#include <hip/hip_bf16.h>
#include <stdint.h>

#define B_  512
#define T_  48
#define N_  62
#define F_  5
#define E_  496
#define H_  512
#define G4  2048   // 4*H
#define NF  310    // N*F
#define KP  320    // K padded to multiple of 32 for WMMA
#define EPSV 1e-5f

typedef __attribute__((ext_vector_type(16))) __bf16 bf16x16;
typedef __attribute__((ext_vector_type(8)))  float  f32x8;
typedef __attribute__((ext_vector_type(4)))  unsigned int u32x4;
typedef unsigned short u16;

__device__ __forceinline__ u16 f2bf(float x) {
  union { float f; unsigned u; } a; a.f = x;
  unsigned r = a.u + 0x7FFFu + ((a.u >> 16) & 1u);   // round-to-nearest-even
  return (u16)(r >> 16);
}
__device__ __forceinline__ float sigf(float x) { return 1.0f / (1.0f + expf(-x)); }

// A fragment (16x32 bf16): lane L -> row m0+(L&15); halves of lanes split K.
// lanes 0-15: K = k0+{0..7, 16..23}; lanes 16-31: K = k0+{8..15, 24..31}.
__device__ __forceinline__ bf16x16 load_frag_a(const u16* row, int k0, int half) {
  union { bf16x16 v; u32x4 u[2]; } r;
  r.u[0] = *reinterpret_cast<const u32x4*>(row + k0 + 8 * half);
  r.u[1] = *reinterpret_cast<const u32x4*>(row + k0 + 16 + 8 * half);
  return r.v;
}
// B fragment (32x16 bf16, K-major storage): lane L -> column N=(L&15); K = {0..7,16..23} + 8*half.
__device__ __forceinline__ bf16x16 load_frag_b(const u16* p, int ld, int half) {
  union { bf16x16 v; u16 e[16]; } r;
#pragma unroll
  for (int s = 0; s < 8; ++s) {
    r.e[s]     = p[(size_t)(8 * half + s) * ld];
    r.e[8 + s] = p[(size_t)(16 + 8 * half + s) * ld];
  }
  return r.v;
}
// generic two-pointer fragment gather (used for LDS reads -> ds_load_b128)
__device__ __forceinline__ bf16x16 load_frag_2(const u16* p0, const u16* p1) {
  union { bf16x16 v; u32x4 u[2]; } r;
  r.u[0] = *reinterpret_cast<const u32x4*>(p0);
  r.u[1] = *reinterpret_cast<const u32x4*>(p1);
  return r.v;
}

// async 16B global -> LDS copy (per-lane addresses; one wave instruction moves 512B)
__device__ __forceinline__ void async_copy_b128(unsigned lds_off, const u16* g) {
  asm volatile("global_load_async_to_lds_b128 %0, %1, off"
               :: "v"(lds_off), "v"((unsigned long long)(uintptr_t)g)
               : "memory");
}

__device__ __forceinline__ void store_tile_16x64(float* __restrict__ D, int m0,
                                                 int ncol, int half,
                                                 const f32x8 acc[4]) {
#pragma unroll
  for (int i = 0; i < 4; ++i) {
#pragma unroll
    for (int e = 0; e < 8; ++e) {
      const int r = m0 + e + 8 * half;
      D[(size_t)r * G4 + ncol + i * 16] = acc[i][e];
    }
  }
}

// ---------------- graph Laplacian ----------------
__global__ __launch_bounds__(64)
void k_build_lhat(const int* __restrict__ esrc, const int* __restrict__ edst,
                  const float* __restrict__ ew, float* __restrict__ Lhat) {
  const int b = blockIdx.x;
  __shared__ float A[N_ * N_];
  __shared__ int ss[E_], dd[E_];
  __shared__ float ww[E_];
  __shared__ float dinv[N_];
  for (int i = threadIdx.x; i < N_ * N_; i += 64) A[i] = 0.f;
  for (int e = threadIdx.x; e < E_; e += 64) {
    ss[e] = esrc[b * E_ + e]; dd[e] = edst[b * E_ + e]; ww[e] = ew[b * E_ + e];
  }
  __syncthreads();
  if (threadIdx.x < N_) {
    const int n = threadIdx.x;
    for (int e = 0; e < E_; ++e)
      if (dd[e] == n) A[n * N_ + ss[e]] += ww[e];        // deterministic order
    float deg = 0.f;
    for (int m = 0; m < N_; ++m) deg += A[n * N_ + m];
    dinv[n] = (deg > 0.f) ? rsqrtf(deg) : 0.f;
  }
  __syncthreads();
  for (int i = threadIdx.x; i < N_ * N_; i += 64) {
    int n = i / N_, m = i % N_;
    Lhat[(size_t)b * (N_ * N_) + i] = -dinv[n] * A[i] * dinv[m];
  }
}

// ---------------- ChebConv K=3 per (t,b) ----------------
__global__ __launch_bounds__(128)
void k_cheb(const float* __restrict__ x, const float* __restrict__ Lhat,
            const float* __restrict__ Wc, const float* __restrict__ bc,
            float* __restrict__ cheb) {
  const int tb = blockIdx.x;
  const int t = tb / B_;
  const int b = tb % B_;
  __shared__ float Lh[N_ * N_];
  __shared__ float X[NF], T1[NF], T2[NF];
  __shared__ float Wsh[80];                       // 3*25 weights + 5 bias
  for (int i = threadIdx.x; i < N_ * N_; i += 128) Lh[i] = Lhat[(size_t)b * (N_ * N_) + i];
  const float* xp = x + ((size_t)b * T_ + t) * NF;
  for (int i = threadIdx.x; i < NF; i += 128) X[i] = xp[i];
  for (int i = threadIdx.x; i < 75; i += 128) Wsh[i] = Wc[t * 75 + i];
  if (threadIdx.x < F_) Wsh[75 + threadIdx.x] = bc[t * F_ + threadIdx.x];
  __syncthreads();
  for (int i = threadIdx.x; i < NF; i += 128) {
    int n = i / F_, f = i % F_;
    float s = 0.f;
    for (int m = 0; m < N_; ++m) s += Lh[n * N_ + m] * X[m * F_ + f];
    T1[i] = s;
  }
  __syncthreads();
  for (int i = threadIdx.x; i < NF; i += 128) {
    int n = i / F_, f = i % F_;
    float s = 0.f;
    for (int m = 0; m < N_; ++m) s += Lh[n * N_ + m] * T1[m * F_ + f];
    T2[i] = 2.f * s - X[i];
  }
  __syncthreads();
  float* op = cheb + ((size_t)t * B_ + b) * NF;
  for (int i = threadIdx.x; i < NF; i += 128) {
    int n = i / F_, g = i % F_;
    float s = Wsh[75 + g];
    for (int f = 0; f < F_; ++f) {
      s += X[n * F_ + f]  * Wsh[f * F_ + g]
         + T1[n * F_ + f] * Wsh[25 + f * F_ + g]
         + T2[n * F_ + f] * Wsh[50 + f * F_ + g];
    }
    op[i] = s;
  }
}

// ---------------- BN over (B,N) per (t,f) ----------------
__global__ __launch_bounds__(256)
void k_bn_stats(const float* __restrict__ cheb, float* __restrict__ mean,
                float* __restrict__ rstd) {
  const int tf = blockIdx.x;
  const int t = tf / F_, f = tf % F_;
  const int cnt = B_ * N_;                         // 31744
  const float* base = cheb + (size_t)t * cnt * F_ + f;
  float s = 0.f, q = 0.f;
  for (int i = threadIdx.x; i < cnt; i += 256) {
    float v = base[(size_t)i * F_];
    s += v; q += v * v;
  }
  __shared__ float ss[256], qq[256];
  ss[threadIdx.x] = s; qq[threadIdx.x] = q; __syncthreads();
  for (int o = 128; o > 0; o >>= 1) {
    if (threadIdx.x < o) { ss[threadIdx.x] += ss[threadIdx.x + o]; qq[threadIdx.x] += qq[threadIdx.x + o]; }
    __syncthreads();
  }
  if (threadIdx.x == 0) {
    float m = ss[0] / (float)cnt;
    float v = qq[0] / (float)cnt - m * m;
    mean[tf] = m;
    rstd[tf] = rsqrtf(fmaxf(v, 0.f) + EPSV);
  }
}

// ---------------- normalized seq -> bf16, K padded ----------------
__global__ __launch_bounds__(256)
void k_make_seq(const float* __restrict__ cheb, const float* __restrict__ mean,
                const float* __restrict__ rstd, const float* __restrict__ gam,
                const float* __restrict__ bet, u16* __restrict__ seqb) {
  const size_t idx = (size_t)blockIdx.x * 256 + threadIdx.x;   // < B*T*KP
  if (idx >= (size_t)B_ * T_ * KP) return;
  const int b = (int)(idx / (T_ * KP));
  const int rem = (int)(idx % (T_ * KP));
  const int t = rem / KP;
  const int kp = rem % KP;
  float v = 0.f;
  if (kp < NF) {
    const int n = kp / F_, f = kp % F_;
    const int tf = t * F_ + f;
    const float raw = cheb[((size_t)t * B_ * N_ + (size_t)b * N_ + n) * F_ + f];
    v = (raw - mean[tf]) * rstd[tf] * gam[tf] + bet[tf];
  }
  seqb[idx] = f2bf(v);
}

// ---------------- weight transposes to K-major bf16 ----------------
__global__ __launch_bounds__(256)
void k_trans_wih(const float* __restrict__ W, u16* __restrict__ Wt) {
  const int idx = blockIdx.x * 256 + threadIdx.x;              // < KP*G4
  if (idx >= KP * G4) return;
  const int k = idx / G4, j = idx % G4;
  Wt[idx] = (k < NF) ? f2bf(W[(size_t)j * NF + k]) : (u16)0;
}
__global__ __launch_bounds__(256)
void k_trans_whh(const float* __restrict__ W, u16* __restrict__ Wt) {
  const int idx = blockIdx.x * 256 + threadIdx.x;              // < H*G4
  if (idx >= H_ * G4) return;
  const int k = idx / G4, j = idx % G4;
  Wt[idx] = f2bf(W[(size_t)j * H_ + k]);
}

__global__ __launch_bounds__(256)
void k_init_state(float* __restrict__ c, u16* __restrict__ h) {
  const int i = blockIdx.x * 256 + threadIdx.x;
  if (i < B_ * H_) { c[i] = 0.f; h[i] = 0; }
}

// ---------------- big input-projection GEMM: (B*T,320) x (320,2048) ----------------
// 49152 waves -> occupancy hides global latency; register-direct fragments.
__global__ __launch_bounds__(256)
void k_gemm_in(const u16* __restrict__ Aseq, const u16* __restrict__ Bt,
               float* __restrict__ D) {
  const int lane = threadIdx.x & 31;
  const int wave = threadIdx.x >> 5;
  const int waveId = blockIdx.x * 8 + wave;        // < 49152
  const int MT = (B_ * T_) / 16;                   // 1536 row tiles
  const int mtile = waveId % MT;
  const int ntile = waveId / MT;                   // 0..31 (64-wide)
  const int m0 = mtile * 16;
  const int n0 = ntile * 64;
  const int half = lane >> 4;
  const u16* arp = Aseq + (size_t)(m0 + (lane & 15)) * KP;
  const u16* bbase = Bt + n0 + (lane & 15);
  const f32x8 fz = {0.f,0.f,0.f,0.f,0.f,0.f,0.f,0.f};
  f32x8 acc[4];
#pragma unroll
  for (int i = 0; i < 4; ++i) acc[i] = fz;
#pragma unroll
  for (int k0 = 0; k0 < KP; k0 += 32) {
    bf16x16 a = load_frag_a(arp, k0, half);
    const u16* bp = bbase + (size_t)k0 * G4;
    if (k0 + 32 < KP) {
      __builtin_prefetch(arp + k0 + 32, 0, 3);
      __builtin_prefetch(bbase + (size_t)(k0 + 32) * G4, 0, 3);
    }
#pragma unroll
    for (int i = 0; i < 4; ++i) {
      bf16x16 b = load_frag_b(bp + i * 16, G4, half);
      acc[i] = __builtin_amdgcn_wmma_f32_16x16x32_bf16(false, a, false, b,
                                                       (short)0, acc[i], false, false);
    }
  }
  store_tile_16x64(D, m0, n0 + (lane & 15), half, acc);
}

// ---------------- recurrent GEMM: (512,512) x (512,2048) ----------------
// Few waves in flight (sequential step) -> hide latency with the async
// global->LDS engine: 2-stage LDS double buffer, s_wait_asynccnt pipeline.
// Block = 8 waves, tile 128(M) x 64(N); stage = 32 K.
#define AROW 40   // A LDS row stride (elements): 32 + 8 pad (80B)
#define BROW 72   // B LDS row stride (elements): 64 + 8 pad (144B)
__global__ __launch_bounds__(256)
void k_gemm_rec(const u16* __restrict__ Ah, const u16* __restrict__ Bt,
                float* __restrict__ D) {
  __shared__ u16 Abuf[2][128 * AROW];
  __shared__ u16 Bbuf[2][32 * BROW];
  const int tid  = threadIdx.x;
  const int lane = tid & 31;
  const int wave = tid >> 5;
  const int half = lane >> 4;
  const int mBlock = (blockIdx.x & 3) * 128;       // 512 / 128
  const int n0     = (blockIdx.x >> 2) * 64;       // 2048 / 64
  const int m0w    = mBlock + wave * 16;

  // per-thread async chunk coordinates (16B chunks)
  const int ar0 = tid >> 2,        ac0 = (tid & 3) * 8;          // A chunk 1: rows 0..63
  const int ar1 = (tid + 256) >> 2, ac1 = (tid & 3) * 8;         // A chunk 2: rows 64..127
  const int br  = tid >> 3,        bcx = (tid & 7) * 8;          // B chunk: k-rows 0..31
  const unsigned aL0 = (unsigned)(uintptr_t)&Abuf[0][ar0 * AROW + ac0];
  const unsigned aL1 = (unsigned)(uintptr_t)&Abuf[0][ar1 * AROW + ac1];
  const unsigned bL  = (unsigned)(uintptr_t)&Bbuf[0][br * BROW + bcx];
  const unsigned aStride = (unsigned)sizeof(Abuf[0]);
  const unsigned bStride = (unsigned)sizeof(Bbuf[0]);
  const u16* ag0 = Ah + (size_t)(mBlock + ar0) * H_ + ac0;
  const u16* ag1 = Ah + (size_t)(mBlock + ar1) * H_ + ac1;
  const u16* bg  = Bt + (size_t)br * G4 + n0 + bcx;

  // prologue: stage 0
  async_copy_b128(aL0, ag0);
  async_copy_b128(aL1, ag1);
  async_copy_b128(bL, bg);

  const f32x8 fz = {0.f,0.f,0.f,0.f,0.f,0.f,0.f,0.f};
  f32x8 acc[4];
#pragma unroll
  for (int i = 0; i < 4; ++i) acc[i] = fz;

#pragma unroll
  for (int ks = 0; ks < H_ / 32; ++ks) {           // 16 stages
    const int buf = ks & 1;
    if (ks + 1 < H_ / 32) {                        // stream next stage
      const int k0 = (ks + 1) * 32;
      const unsigned off = (unsigned)((ks + 1) & 1);
      async_copy_b128(aL0 + off * aStride, ag0 + k0);
      async_copy_b128(aL1 + off * aStride, ag1 + k0);
      async_copy_b128(bL + off * bStride, bg + (size_t)k0 * G4);
      asm volatile("s_wait_asynccnt 0x3" ::: "memory");  // stage ks landed
    } else {
      asm volatile("s_wait_asynccnt 0x0" ::: "memory");
    }
    __syncthreads();                               // all waves' stage data visible
    const u16* arow = &Abuf[buf][(wave * 16 + (lane & 15)) * AROW];
    bf16x16 a = load_frag_2(arow + 8 * half, arow + 16 + 8 * half);
    const u16* brow = &Bbuf[buf][lane & 15];
#pragma unroll
    for (int i = 0; i < 4; ++i) {
      bf16x16 b = load_frag_b(brow + i * 16, BROW, half);
      acc[i] = __builtin_amdgcn_wmma_f32_16x16x32_bf16(false, a, false, b,
                                                       (short)0, acc[i], false, false);
    }
    __syncthreads();                               // buffer free for stage ks+2
  }
  store_tile_16x64(D, m0w, n0 + (lane & 15), half, acc);
}

// ---------------- LSTM gates (i,f,g,o) ----------------
__global__ __launch_bounds__(256)
void k_lstm_gate(const float* __restrict__ Gin, const float* __restrict__ Ghat,
                 const float* __restrict__ b_ih, const float* __restrict__ b_hh,
                 float* __restrict__ c, u16* __restrict__ hbf,
                 float* __restrict__ y, int t) {
  const int i = blockIdx.x * 256 + threadIdx.x;    // < B*H
  if (i >= B_ * H_) return;
  const int b = i >> 9;
  const int hh = i & (H_ - 1);
  const size_t gb = ((size_t)b * T_ + t) * G4;
  const size_t hb = (size_t)b * G4;
  const float gi = Gin[gb + hh]            + Ghat[hb + hh]            + b_ih[hh]            + b_hh[hh];
  const float gf = Gin[gb + H_ + hh]       + Ghat[hb + H_ + hh]       + b_ih[H_ + hh]       + b_hh[H_ + hh];
  const float gg = Gin[gb + 2 * H_ + hh]   + Ghat[hb + 2 * H_ + hh]   + b_ih[2 * H_ + hh]   + b_hh[2 * H_ + hh];
  const float go = Gin[gb + 3 * H_ + hh]   + Ghat[hb + 3 * H_ + hh]   + b_ih[3 * H_ + hh]   + b_hh[3 * H_ + hh];
  const float cp = c[i];
  const float cn = sigf(gf) * cp + sigf(gi) * tanhf(gg);
  const float hn = sigf(go) * tanhf(cn);
  c[i] = cn;
  hbf[i] = f2bf(hn);
  y[((size_t)b * T_ + t) * H_ + hh] = hn;
}

// ---------------- BN1 over (B,H) per t ----------------
__global__ __launch_bounds__(256)
void k_bn1_stats(const float* __restrict__ y, float* __restrict__ mean,
                 float* __restrict__ rstd) {
  const int t = blockIdx.x;
  const int cnt = B_ * H_;                          // 262144
  float s = 0.f, q = 0.f;
  for (int i = threadIdx.x; i < cnt; i += 256) {
    const int b = i >> 9, hh = i & (H_ - 1);
    const float v = y[((size_t)b * T_ + t) * H_ + hh];
    s += v; q += v * v;
  }
  __shared__ float ss[256], qq[256];
  ss[threadIdx.x] = s; qq[threadIdx.x] = q; __syncthreads();
  for (int o = 128; o > 0; o >>= 1) {
    if (threadIdx.x < o) { ss[threadIdx.x] += ss[threadIdx.x + o]; qq[threadIdx.x] += qq[threadIdx.x + o]; }
    __syncthreads();
  }
  if (threadIdx.x == 0) {
    const float m = ss[0] / (float)cnt;
    const float v = qq[0] / (float)cnt - m * m;
    mean[t] = m;
    rstd[t] = rsqrtf(fmaxf(v, 0.f) + EPSV);
  }
}

// ---------------- head: (B, T*H) x (T*H, 3), BN1 folded in ----------------
__global__ __launch_bounds__(256)
void k_head(const float* __restrict__ y, const float* __restrict__ mu1,
            const float* __restrict__ rs1, const float* __restrict__ g1,
            const float* __restrict__ be1, const float* __restrict__ Wl,
            const float* __restrict__ bl, float* __restrict__ out) {
  const int b = blockIdx.x;
  const int TH = T_ * H_;                           // 24576
  float a0 = 0.f, a1 = 0.f, a2 = 0.f;
  for (int idx = threadIdx.x; idx < TH; idx += 256) {
    const int t = idx >> 9, hh = idx & (H_ - 1);
    const float v = (y[((size_t)b * T_ + t) * H_ + hh] - mu1[t]) * rs1[t] * g1[t] + be1[t];
    a0 += v * Wl[idx];
    a1 += v * Wl[TH + idx];
    a2 += v * Wl[2 * TH + idx];
  }
  __shared__ float s0[256], s1[256], s2[256];
  s0[threadIdx.x] = a0; s1[threadIdx.x] = a1; s2[threadIdx.x] = a2; __syncthreads();
  for (int o = 128; o > 0; o >>= 1) {
    if (threadIdx.x < o) {
      s0[threadIdx.x] += s0[threadIdx.x + o];
      s1[threadIdx.x] += s1[threadIdx.x + o];
      s2[threadIdx.x] += s2[threadIdx.x + o];
    }
    __syncthreads();
  }
  if (threadIdx.x == 0) {
    out[b * 3 + 0] = s0[0] + bl[0];
    out[b * 3 + 1] = s1[0] + bl[1];
    out[b * 3 + 2] = s2[0] + bl[2];
  }
}

extern "C" void kernel_launch(void* const* d_in, const int* in_sizes, int n_in,
                              void* d_out, int out_size, void* d_ws, size_t ws_size,
                              hipStream_t stream) {
  (void)in_sizes; (void)n_in; (void)out_size; (void)ws_size;
  const float* x     = (const float*)d_in[0];
  const int*   esrc  = (const int*)d_in[1];
  const int*   edst  = (const int*)d_in[2];
  const float* ew    = (const float*)d_in[3];
  const float* Wc    = (const float*)d_in[4];
  const float* bc    = (const float*)d_in[5];
  const float* bng   = (const float*)d_in[6];
  const float* bnb   = (const float*)d_in[7];
  const float* W_ih  = (const float*)d_in[8];
  const float* W_hh  = (const float*)d_in[9];
  const float* b_ih  = (const float*)d_in[10];
  const float* b_hh  = (const float*)d_in[11];
  const float* bn1g  = (const float*)d_in[12];
  const float* bn1b  = (const float*)d_in[13];
  const float* Wl    = (const float*)d_in[14];
  const float* bl    = (const float*)d_in[15];
  float* out = (float*)d_out;

  char* p = (char*)d_ws;
  auto take = [&](size_t bytes) -> char* {
    char* r = p; p += (bytes + 255) & ~(size_t)255; return r;
  };
  float* Lhat = (float*)take((size_t)B_ * N_ * N_ * 4);
  float* cheb = (float*)take((size_t)T_ * B_ * NF * 4);
  float* bn_m = (float*)take((size_t)T_ * F_ * 4);
  float* bn_r = (float*)take((size_t)T_ * F_ * 4);
  u16*   seqb = (u16*)take((size_t)B_ * T_ * KP * 2);
  u16*   WihT = (u16*)take((size_t)KP * G4 * 2);
  u16*   WhhT = (u16*)take((size_t)H_ * G4 * 2);
  float* Gin  = (float*)take((size_t)B_ * T_ * G4 * 4);
  float* Ghat = (float*)take((size_t)B_ * G4 * 4);
  float* cst  = (float*)take((size_t)B_ * H_ * 4);
  u16*   hbf  = (u16*)take((size_t)B_ * H_ * 2);
  float* y    = (float*)take((size_t)B_ * T_ * H_ * 4);
  float* bn1m = (float*)take((size_t)T_ * 4);
  float* bn1r = (float*)take((size_t)T_ * 4);

  // state + weight prep
  k_init_state<<<(B_ * H_ + 255) / 256, 256, 0, stream>>>(cst, hbf);
  k_build_lhat<<<B_, 64, 0, stream>>>(esrc, edst, ew, Lhat);
  k_trans_wih<<<(KP * G4 + 255) / 256, 256, 0, stream>>>(W_ih, WihT);
  k_trans_whh<<<(H_ * G4 + 255) / 256, 256, 0, stream>>>(W_hh, WhhT);

  // graph conv + BN + sequence build
  k_cheb<<<T_ * B_, 128, 0, stream>>>(x, Lhat, Wc, bc, cheb);
  k_bn_stats<<<T_ * F_, 256, 0, stream>>>(cheb, bn_m, bn_r);
  {
    const size_t n = (size_t)B_ * T_ * KP;
    k_make_seq<<<(unsigned)((n + 255) / 256), 256, 0, stream>>>(cheb, bn_m, bn_r, bng, bnb, seqb);
  }

  // all input projections at once: 24576 x 2048 x 320 (WMMA bf16)
  k_gemm_in<<<((B_ * T_ / 16) * (G4 / 64)) / 8, 256, 0, stream>>>(seqb, WihT, Gin);

  // sequential LSTM recurrence: 48 x (512 x 2048 x 512 WMMA GEMM + gates)
  for (int t = 0; t < T_; ++t) {
    k_gemm_rec<<<(512 / 128) * (G4 / 64), 256, 0, stream>>>(hbf, WhhT, Ghat);
    k_lstm_gate<<<(B_ * H_) / 256, 256, 0, stream>>>(Gin, Ghat, b_ih, b_hh, cst, hbf, y, t);
  }

  // BN1 + head
  k_bn1_stats<<<T_, 256, 0, stream>>>(y, bn1m, bn1r);
  k_head<<<B_, 256, 0, stream>>>(y, bn1m, bn1r, bn1g, bn1b, Wl, bl, out);
}
